// SparseGATLayer_1580547975120
// MI455X (gfx1250) — matrix operand order repaired and checked by reading, not compiled, not run on hardware
//
#include <hip/hip_runtime.h>
#include <hip/hip_bf16.h>
#include <stdint.h>

// ---------------- WMMA types ----------------
typedef __attribute__((ext_vector_type(16))) __bf16 v16bf;
typedef __attribute__((ext_vector_type(8)))  float  v8f;

union FragBF { unsigned u[8]; v16bf v; };

__device__ __forceinline__ unsigned pkbf(float a, float b) {
    __bf16 lo = (__bf16)a, hi = (__bf16)b;
    unsigned short ulo = __builtin_bit_cast(unsigned short, lo);
    unsigned short uhi = __builtin_bit_cast(unsigned short, hi);
    return (unsigned)ulo | ((unsigned)uhi << 16);
}

// monotonic float <-> uint key for atomic scatter-max
__device__ __forceinline__ unsigned f2key(float f) {
    unsigned i = __float_as_uint(f);
    return (i & 0x80000000u) ? ~i : (i | 0x80000000u);
}
__device__ __forceinline__ float key2f(unsigned k) {
    unsigned i = (k & 0x80000000u) ? (k & 0x7fffffffu) : ~k;
    return __uint_as_float(i);
}

#define BLK    128   // 4 waves of 32
#define ROWS   128   // node rows per block (32 per wave)
#define PITCH  68    // packed-uint pitch (bank-skewed, 16B aligned)

// =====================================================================
// K1: q,k,v = x @ W{q,k,v} via bf16 WMMA (f32 accumulate).
// Stores v (N x 128) plus the per-node reductions:
//   score_src, score_dst (N x 4),  qe[n][h][t] = q[n,h,:] . ee[t,h,:]
// Each wave owns 32 rows: 2 A-fragment sets share each B fragment.
// =====================================================================
__global__ void __launch_bounds__(BLK)
qkv_score_kernel(const float* __restrict__ x,
                 const float* __restrict__ Wq,
                 const float* __restrict__ Wk,
                 const float* __restrict__ Wv,
                 const float* __restrict__ attn,   // 256 = H x 2D
                 const float* __restrict__ ee,     // 384 = 3 x 128
                 float* __restrict__ vout,
                 float* __restrict__ ssrc,
                 float* __restrict__ sdst,
                 float* __restrict__ qe,
                 int N)
{
    // buf holds the x stripe first (dead after A-frag load), then W^T.
    __shared__ unsigned buf[128 * PITCH];          // 34816 B
    __shared__ float accS[ROWS * 4];
    __shared__ float accD[ROWS * 4];
    __shared__ float accQ[ROWS * 12];
    __shared__ float sAttn[256];
    __shared__ float sEE[384];

    const int tid  = threadIdx.x;
    const int row0 = blockIdx.x * ROWS;

    for (int i = tid; i < 256; i += BLK) sAttn[i] = attn[i];
    for (int i = tid; i < 384; i += BLK) sEE[i]   = ee[i];
    for (int i = tid; i < ROWS * 4;  i += BLK) { accS[i] = 0.f; accD[i] = 0.f; }
    for (int i = tid; i < ROWS * 12; i += BLK) accQ[i] = 0.f;

    // stage x stripe as packed bf16 pairs (row r, K-pair kp)
    for (int it = 0; it < 64; ++it) {
        int i = it * BLK + tid;
        int r = i >> 6, kp = i & 63;
        int g = row0 + r;
        float a = 0.f, b = 0.f;
        if (g < N) { const float* p = x + (size_t)g * 128 + kp * 2; a = p[0]; b = p[1]; }
        buf[r * PITCH + kp] = pkbf(a, b);
    }
    __syncthreads();

    const int lane = tid & 31;
    const int wave = tid >> 5;         // 0..3
    const int kh   = lane >> 4;        // K-half select
    const int m15  = lane & 15;

    // A fragments: 2 row-sets x 4 K-chunks, persistent in registers
    FragBF afrag[2][4];
#pragma unroll
    for (int s = 0; s < 2; ++s) {
        int arow = wave * 32 + s * 16 + m15;
#pragma unroll
        for (int c = 0; c < 4; ++c) {
            int base = arow * PITCH + c * 16 + kh * 8;
#pragma unroll
            for (int j = 0; j < 4; ++j) afrag[s][c].u[j]     = buf[base + j];
#pragma unroll
            for (int j = 0; j < 4; ++j) afrag[s][c].u[4 + j] = buf[base + 4 + j];
        }
    }
    __syncthreads();   // x stripe dead; buf can be reused for W^T

#pragma unroll
    for (int p = 0; p < 3; ++p) {
        const float* W = (p == 0) ? Wq : ((p == 1) ? Wk : Wv);
        // stage W^T packed: buf[n * PITCH + kp] holds W[2kp][n], W[2kp+1][n]
        for (int it = 0; it < 64; ++it) {
            float a = W[(2 * it) * 128 + tid];
            float b = W[(2 * it + 1) * 128 + tid];
            buf[tid * PITCH + it] = pkbf(a, b);
        }
        __syncthreads();

        for (int nt = 0; nt < 8; ++nt) {
            const int ncol = nt * 16 + m15;
            // load all 4 B fragments up front (overlappable ds_load_b128s)
            FragBF bf[4];
#pragma unroll
            for (int cc = 0; cc < 4; ++cc) {
                int base = ncol * PITCH + cc * 16 + kh * 8;
#pragma unroll
                for (int j = 0; j < 8; ++j) bf[cc].u[j] = buf[base + j];
            }
            v8f z = {0.f,0.f,0.f,0.f,0.f,0.f,0.f,0.f};
            v8f cacc[2]; cacc[0] = z; cacc[1] = z;
#pragma unroll
            for (int cc = 0; cc < 4; ++cc) {
                cacc[0] = __builtin_amdgcn_wmma_f32_16x16x32_bf16(
                    false, afrag[0][cc].v, false, bf[cc].v, (short)0, cacc[0], false, false);
                cacc[1] = __builtin_amdgcn_wmma_f32_16x16x32_bf16(
                    false, afrag[1][cc].v, false, bf[cc].v, (short)0, cacc[1], false, false);
            }

            if (p == 2) {
                // store v tiles
#pragma unroll
                for (int s = 0; s < 2; ++s)
#pragma unroll
                for (int j = 0; j < 8; ++j) {
                    int g = row0 + wave * 32 + s * 16 + kh * 8 + j;
                    if (g < N) vout[(size_t)g * 128 + ncol] = cacc[s][j];
                }
            } else {
                const int h = ncol >> 5, d = ncol & 31;
                const float al = (p == 0) ? sAttn[h * 64 + d] : sAttn[h * 64 + 32 + d];
                const float e0 = sEE[ncol], e1 = sEE[128 + ncol], e2 = sEE[256 + ncol];
#pragma unroll
                for (int s = 0; s < 2; ++s)
#pragma unroll
                for (int j = 0; j < 8; ++j) {
                    int ml = wave * 32 + s * 16 + kh * 8 + j;
                    float qv = cacc[s][j];
                    if (p == 0) {
                        atomicAdd(&accS[ml * 4 + h], qv * al);
                        atomicAdd(&accQ[ml * 12 + h * 3 + 0], qv * e0);
                        atomicAdd(&accQ[ml * 12 + h * 3 + 1], qv * e1);
                        atomicAdd(&accQ[ml * 12 + h * 3 + 2], qv * e2);
                    } else {
                        atomicAdd(&accD[ml * 4 + h], qv * al);
                    }
                }
            }
        }
        __syncthreads();   // done reading buf for this phase
    }

    for (int i = tid; i < ROWS * 4; i += BLK) {
        int g = row0 + (i >> 2);
        if (g < N) { ssrc[(size_t)g * 4 + (i & 3)] = accS[i];
                     sdst[(size_t)g * 4 + (i & 3)] = accD[i]; }
    }
    for (int i = tid; i < ROWS * 12; i += BLK) {
        int g = row0 + (i / 12);
        if (g < N) qe[(size_t)g * 12 + (i % 12)] = accQ[i];
    }
}

// =====================================================================
// K_init: max_s <- key(-100), sum_exp <- 0, accum <- 0
// =====================================================================
__global__ void init_kernel(unsigned* __restrict__ maxs,
                            float* __restrict__ sumexp,
                            float* __restrict__ accum,
                            int nMax, int nAcc)
{
    const unsigned negk = f2key(-100.0f);
    int i = blockIdx.x * blockDim.x + threadIdx.x;
    int stride = gridDim.x * blockDim.x;
    for (; i < nAcc; i += stride) {
        if (i < nMax) { maxs[i] = negk; sumexp[i] = 0.f; }
        accum[i] = 0.f;
    }
}

// =====================================================================
// K2: per-edge raw scores + scatter-max (uint key). Also prefetch v[src]
// rows toward L2 for the later message pass.
// =====================================================================
__global__ void edge_score_kernel(const int* __restrict__ ei,
                                  const int* __restrict__ et,
                                  const float* __restrict__ ssrc,
                                  const float* __restrict__ sdst,
                                  const float* __restrict__ qe,
                                  const float* __restrict__ v,
                                  float* __restrict__ sbuf,
                                  unsigned* __restrict__ maxs,
                                  int E)
{
    int e = blockIdx.x * blockDim.x + threadIdx.x;
    if (e >= E) return;
    int s = ei[e], d = ei[E + e], t = et[e];
    __builtin_prefetch(v + (size_t)s * 128, 0, 1);   // global_prefetch: warm v[src]
#pragma unroll
    for (int h = 0; h < 4; ++h) {
        float sv = ssrc[(size_t)s * 4 + h] + sdst[(size_t)d * 4 + h];
        sv = (sv >= 0.f) ? sv : 0.2f * sv;                 // leaky relu
        sv += qe[(size_t)s * 12 + h * 3 + t];
        sbuf[(size_t)e * 4 + h] = sv;
        atomicMax(&maxs[(size_t)d * 4 + h], f2key(sv));
    }
}

// =====================================================================
// K3: exp(s - max) in place + scatter-sum
// =====================================================================
__global__ void edge_exp_kernel(const int* __restrict__ ei,
                                const unsigned* __restrict__ maxs,
                                float* __restrict__ sbuf,
                                float* __restrict__ sumexp,
                                int E)
{
    int e = blockIdx.x * blockDim.x + threadIdx.x;
    if (e >= E) return;
    int d = ei[E + e];
#pragma unroll
    for (int h = 0; h < 4; ++h) {
        float ms = key2f(maxs[(size_t)d * 4 + h]);
        float es = __expf(sbuf[(size_t)e * 4 + h] - ms);
        sbuf[(size_t)e * 4 + h] = es;
        atomicAdd(&sumexp[(size_t)d * 4 + h], es);
    }
}

// =====================================================================
// K4: messages. 32 threads per edge (one D-lane each), 4 heads per thread.
// =====================================================================
__global__ void edge_msg_kernel(const int* __restrict__ ei,
                                const float* __restrict__ sbuf,
                                const float* __restrict__ sumexp,
                                const float* __restrict__ v,
                                float* __restrict__ accum,
                                int E)
{
    int id = blockIdx.x * blockDim.x + threadIdx.x;
    if (id >= E * 32) return;
    int e  = id >> 5;
    int d0 = id & 31;
    int s = ei[e], d = ei[E + e];
#pragma unroll
    for (int h = 0; h < 4; ++h) {
        float a = fminf(sbuf[(size_t)e * 4 + h] /
                        (sumexp[(size_t)d * 4 + h] + 1e-6f), 1.0f);
        atomicAdd(&accum[(size_t)d * 128 + h * 32 + d0],
                  a * v[(size_t)s * 128 + h * 32 + d0]);
    }
}

// =====================================================================
// K5: out = clip(accum,-50,50) @ Wo + bo  (bf16 WMMA, f32 accumulate)
// =====================================================================
__global__ void __launch_bounds__(BLK)
out_gemm_kernel(const float* __restrict__ acc,
                const float* __restrict__ Wo,
                const float* __restrict__ bo,
                float* __restrict__ out,
                int N)
{
    __shared__ unsigned buf[128 * PITCH];
    __shared__ float sB[128];

    const int tid  = threadIdx.x;
    const int row0 = blockIdx.x * ROWS;

    if (tid < 128) sB[tid] = bo[tid];

    for (int it = 0; it < 64; ++it) {
        int i = it * BLK + tid;
        int r = i >> 6, kp = i & 63;
        int g = row0 + r;
        float a = 0.f, b = 0.f;
        if (g < N) {
            const float* p = acc + (size_t)g * 128 + kp * 2;
            a = fminf(fmaxf(p[0], -50.f), 50.f);
            b = fminf(fmaxf(p[1], -50.f), 50.f);
        }
        buf[r * PITCH + kp] = pkbf(a, b);
    }
    __syncthreads();

    const int lane = tid & 31;
    const int wave = tid >> 5;
    const int kh   = lane >> 4;
    const int m15  = lane & 15;

    FragBF afrag[2][4];
#pragma unroll
    for (int s = 0; s < 2; ++s) {
        int arow = wave * 32 + s * 16 + m15;
#pragma unroll
        for (int c = 0; c < 4; ++c) {
            int base = arow * PITCH + c * 16 + kh * 8;
#pragma unroll
            for (int j = 0; j < 4; ++j) afrag[s][c].u[j]     = buf[base + j];
#pragma unroll
            for (int j = 0; j < 4; ++j) afrag[s][c].u[4 + j] = buf[base + 4 + j];
        }
    }
    __syncthreads();

    for (int it = 0; it < 64; ++it) {
        float a = Wo[(2 * it) * 128 + tid];
        float b = Wo[(2 * it + 1) * 128 + tid];
        buf[tid * PITCH + it] = pkbf(a, b);
    }
    __syncthreads();

    for (int nt = 0; nt < 8; ++nt) {
        const int ncol = nt * 16 + m15;
        FragBF bf[4];
#pragma unroll
        for (int cc = 0; cc < 4; ++cc) {
            int base = ncol * PITCH + cc * 16 + kh * 8;
#pragma unroll
            for (int j = 0; j < 8; ++j) bf[cc].u[j] = buf[base + j];
        }
        v8f z = {0.f,0.f,0.f,0.f,0.f,0.f,0.f,0.f};
        v8f cacc[2]; cacc[0] = z; cacc[1] = z;
#pragma unroll
        for (int cc = 0; cc < 4; ++cc) {
            cacc[0] = __builtin_amdgcn_wmma_f32_16x16x32_bf16(
                false, afrag[0][cc].v, false, bf[cc].v, (short)0, cacc[0], false, false);
            cacc[1] = __builtin_amdgcn_wmma_f32_16x16x32_bf16(
                false, afrag[1][cc].v, false, bf[cc].v, (short)0, cacc[1], false, false);
        }
        float bias = sB[ncol];
#pragma unroll
        for (int s = 0; s < 2; ++s)
#pragma unroll
        for (int j = 0; j < 8; ++j) {
            int g = row0 + wave * 32 + s * 16 + kh * 8 + j;
            if (g < N) out[(size_t)g * 128 + ncol] = cacc[s][j] + bias;
        }
    }
}

// =====================================================================
extern "C" void kernel_launch(void* const* d_in, const int* in_sizes, int n_in,
                              void* d_out, int out_size, void* d_ws, size_t ws_size,
                              hipStream_t stream)
{
    const float* x    = (const float*)d_in[0];
    const int*   ei   = (const int*)  d_in[1];
    const int*   et   = (const int*)  d_in[2];
    const float* Wq   = (const float*)d_in[3];
    const float* Wk   = (const float*)d_in[4];
    const float* Wv   = (const float*)d_in[5];
    const float* attn = (const float*)d_in[6];
    const float* ee   = (const float*)d_in[7];
    const float* Wo   = (const float*)d_in[8];
    const float* bo   = (const float*)d_in[9];
    float* out = (float*)d_out;

    const int N = in_sizes[0] / 128;
    const int E = in_sizes[2];

    // workspace layout
    char* ws = (char*)d_ws;
    auto take = [&](size_t bytes) -> char* {
        char* p = ws;
        ws += (bytes + 255) & ~(size_t)255;
        return p;
    };
    float*    vout   = (float*)   take((size_t)N * 128 * 4);
    float*    ssrc   = (float*)   take((size_t)N * 4 * 4);
    float*    sdst   = (float*)   take((size_t)N * 4 * 4);
    float*    qe     = (float*)   take((size_t)N * 12 * 4);
    unsigned* maxs   = (unsigned*)take((size_t)N * 4 * 4);
    float*    sumexp = (float*)   take((size_t)N * 4 * 4);
    float*    sbuf   = (float*)   take((size_t)E * 4 * 4);
    float*    accum  = (float*)   take((size_t)N * 128 * 4);

    const int nBlocksRows = (N + ROWS - 1) / ROWS;

    qkv_score_kernel<<<nBlocksRows, BLK, 0, stream>>>(
        x, Wq, Wk, Wv, attn, ee, vout, ssrc, sdst, qe, N);

    {
        int nAcc = N * 128, nMax = N * 4;
        int blocks = (nAcc + 255) / 256;
        if (blocks > 32768) blocks = 32768;
        init_kernel<<<blocks, 256, 0, stream>>>(maxs, sumexp, accum, nMax, nAcc);
    }

    edge_score_kernel<<<(E + 255) / 256, 256, 0, stream>>>(
        ei, et, ssrc, sdst, qe, vout, sbuf, maxs, E);

    edge_exp_kernel<<<(E + 255) / 256, 256, 0, stream>>>(
        ei, maxs, sbuf, sumexp, E);

    edge_msg_kernel<<<(E * 32 + 255) / 256, 256, 0, stream>>>(
        ei, sbuf, sumexp, vout, accum, E);

    out_gemm_kernel<<<nBlocksRows, BLK, 0, stream>>>(accum, Wo, bo, out, N);
}
